// MultiHeadAttention_49606872268927
// MI455X (gfx1250) — matrix operand order repaired and checked by reading, not compiled, not run on hardware
//
#include <hip/hip_runtime.h>
#include <hip/hip_bf16.h>
#include <math.h>

// ---------------------------------------------------------------------------
// MHA forward for MI455X (gfx1250, wave32, WMMA, async-LDS).
//   P0: f32 -> bf16 of query/key;  W -> bf16 TRANSPOSED (L2-resident, 2MB)
//   P1: row masks
//   P2: bf16 WMMA GEMM projections, zero LDS (B operand contiguous from Wt);
//       V projection stores directly in transposed per-slice layout vT[d][lk]
//   P3: flash attention; K chunks staged to LDS via global_load_async_to_lds,
//       V columns read contiguously from vT; P shuffle via per-wave LDS tile
//   P4: concat + residual + L2 normalize
// ---------------------------------------------------------------------------

typedef __attribute__((ext_vector_type(16))) __bf16 v16bf;
typedef __attribute__((ext_vector_type(8)))  __bf16 v8bf;
typedef __attribute__((ext_vector_type(8)))  float  v8f;
typedef __attribute__((ext_vector_type(4)))  int    v4i;

#define NEGC (-4294967295.0f)

static __device__ __forceinline__ __bf16 f2bf(float f) { return (__bf16)f; }

static __device__ __forceinline__ v8f wmma_bf16(v16bf a, v16bf b, v8f c) {
  return __builtin_amdgcn_wmma_f32_16x16x32_bf16(
      false, a, false, b, (short)0, c, false, false);
}

union Frag { v16bf v; v8bf h[2]; };

// ---- gfx1250 async global->LDS copy (guarded) ------------------------------
#if __has_builtin(__builtin_amdgcn_global_load_async_to_lds_b128)
#define HAVE_ASYNC_LDS 1
#define AS1 __attribute__((address_space(1)))
#define AS3 __attribute__((address_space(3)))
static __device__ __forceinline__ void async_copy16(const __bf16* gsrc, __bf16* ldst) {
  __builtin_amdgcn_global_load_async_to_lds_b128(
      (AS1 v4i*)(void*)gsrc, (AS3 v4i*)(void*)ldst, 0, 0);
}
static __device__ __forceinline__ void wait_async0() {
#if __has_builtin(__builtin_amdgcn_s_wait_asynccnt)
  __builtin_amdgcn_s_wait_asynccnt(0);
#else
  asm volatile("s_wait_asynccnt 0" ::: "memory");
#endif
}
#else
#define HAVE_ASYNC_LDS 0
#endif

// ---------------------------------------------------------------------------
// P0a: f32 -> bf16 (activations)
// ---------------------------------------------------------------------------
__global__ void k_tobf16(const float* __restrict__ in, __bf16* __restrict__ out, int n) {
  int i = blockIdx.x * blockDim.x + threadIdx.x;
  int stride = gridDim.x * blockDim.x;
  for (; i < n; i += stride) out[i] = f2bf(in[i]);
}

// ---------------------------------------------------------------------------
// P0b: W [1024(k) x 1024(n)] f32 -> Wt [n][k] bf16, LDS-tiled 64x64 transpose
// grid (16,16), block 256
// ---------------------------------------------------------------------------
__global__ void k_wtrans(const float* __restrict__ W, __bf16* __restrict__ Wt) {
  __shared__ __bf16 t[64][65];
  const int tid = threadIdx.x;
  const int bn = blockIdx.x * 64, bk = blockIdx.y * 64;
#pragma unroll
  for (int it = 0; it < 16; ++it) {
    int idx = it * 256 + tid;
    int kk = idx >> 6, nn = idx & 63;               // load coalesced over n
    t[nn][kk] = f2bf(W[(size_t)(bk + kk) * 1024 + bn + nn]);
  }
  __syncthreads();
#pragma unroll
  for (int it = 0; it < 16; ++it) {
    int idx = it * 256 + tid;
    int nn = idx >> 6, kk = idx & 63;               // store coalesced over k
    Wt[(size_t)(bn + nn) * 1024 + bk + kk] = t[nn][kk];
  }
}

// ---------------------------------------------------------------------------
// P1: row masks: dst[row] = (rowsum != 0) ? 1 : 0 ; one wave per row
// ---------------------------------------------------------------------------
__global__ void k_rowmask(const float* __restrict__ src, float* __restrict__ dst, int rows) {
  int gw = blockIdx.x * 8 + (threadIdx.x >> 5);
  int lane = threadIdx.x & 31;
  if (gw >= rows) return;
  const float* r = src + (size_t)gw * 1024;
  float s = 0.f;
#pragma unroll
  for (int e = 0; e < 32; ++e) s += r[lane + e * 32];
#pragma unroll
  for (int m = 16; m >= 1; m >>= 1) s += __shfl_xor(s, m, 32);
  if (lane == 0) dst[gw] = (s != 0.f) ? 1.f : 0.f;
}

// ---------------------------------------------------------------------------
// P2: C = A[4096,1024] @ W  using transposed weights Wt[n][k] (bf16).
// block = 128 (4 waves); block tile M=64,N=64; wave = 16x64. No LDS.
// vmode=1: scatter output into vT[slice=m>>6][d=c&63][lk=(m&63)*16 + (c>>6)]
// ---------------------------------------------------------------------------
__global__ void k_gemm_proj(const __bf16* __restrict__ A, const __bf16* __restrict__ Wt,
                            __bf16* __restrict__ C, int vmode) {
  const int K = 1024, N = 1024;
  const int tid = threadIdx.x, lane = tid & 31, wid = tid >> 5;
  const int l16 = lane & 15;
  const int half8 = (lane & 16) ? 8 : 0;
  const int kb16 = (lane & 16) ? 16 : 0;
  const int n0 = blockIdx.x * 64, m0 = blockIdx.y * 64;
  const int mrow = m0 + wid * 16 + l16;

  v8f acc[4];
#pragma unroll
  for (int t = 0; t < 4; ++t)
#pragma unroll
    for (int g = 0; g < 8; ++g) acc[t][g] = 0.f;

  for (int k0 = 0; k0 < K; k0 += 32) {
    Frag af;
    af.h[0] = *(const v8bf*)(A + (size_t)mrow * K + k0 + kb16);
    af.h[1] = *(const v8bf*)(A + (size_t)mrow * K + k0 + kb16 + 8);
    if (k0 + 32 < K) __builtin_prefetch(A + (size_t)mrow * K + k0 + 32, 0, 0);
#pragma unroll
    for (int t = 0; t < 4; ++t) {
      v16bf b = *(const v16bf*)(Wt + (size_t)(n0 + t * 16 + l16) * K + k0 + kb16);
      acc[t] = wmma_bf16(af.v, b, acc[t]);
    }
  }

#pragma unroll
  for (int t = 0; t < 4; ++t)
#pragma unroll
    for (int g = 0; g < 8; ++g) {
      const int m = m0 + wid * 16 + g + half8;
      const int c = n0 + t * 16 + l16;
      const __bf16 val = f2bf(acc[t][g]);
      if (!vmode) {
        C[(size_t)m * N + c] = val;
      } else {
        const int slice = m >> 6;
        const int lk = ((m & 63) << 4) + (c >> 6);
        const int d = c & 63;
        C[(size_t)slice * 65536 + (size_t)d * 1024 + lk] = val;
      }
    }
}

// ---------------------------------------------------------------------------
// P3: flash attention over one contiguous head-slice.
// grid = (16 qgroups, 64 slices), block = 128 (4 waves), wave = 16 q-rows.
// K chunk (32x64, 4KB contiguous) staged to LDS via async copy, shared by
// all 4 waves. V columns read contiguously from vT. Online softmax with
// half-wave shfl reductions; P goes C-layout -> A-layout via per-wave LDS.
// ---------------------------------------------------------------------------
__global__ void k_attn(const __bf16* __restrict__ Qp, const __bf16* __restrict__ Kp,
                       const __bf16* __restrict__ VTp, const float* __restrict__ kmaskA,
                       const float* __restrict__ qmaskA, float* __restrict__ Oacc) {
  const int D = 64, L = 1024;
  __shared__ __bf16 kt[32 * 64];          // K chunk, row-major [kk][d]
  __shared__ __bf16 pt[4][16 * 32];       // per-wave P tile [row][key]

  const int tid = threadIdx.x, lane = tid & 31, wid = tid >> 5;
  const int l16 = lane & 15;
  const int half8 = (lane & 16) ? 8 : 0;
  const int kb16 = (lane & 16) ? 16 : 0;
  const int i = blockIdx.y;
  const int tq = blockIdx.x * 4 + wid;

  const __bf16* Q  = Qp  + (size_t)i * L * D;
  const __bf16* Kh = Kp  + (size_t)i * L * D;
  const __bf16* VT = VTp + (size_t)i * L * D;   // [64 d][1024 lk]
  const float*  km = kmaskA + (size_t)(i & 3) * L;
  const float*  qm = qmaskA + (size_t)(i & 3) * L;

  const int qrow = tq * 16 + l16;
  Frag qa0, qa1;
  qa0.h[0] = *(const v8bf*)(Q + (size_t)qrow * D + kb16);
  qa0.h[1] = *(const v8bf*)(Q + (size_t)qrow * D + kb16 + 8);
  qa1.h[0] = *(const v8bf*)(Q + (size_t)qrow * D + 32 + kb16);
  qa1.h[1] = *(const v8bf*)(Q + (size_t)qrow * D + 32 + kb16 + 8);

  v8f o[4];
  float mrun[8], lrun[8];
#pragma unroll
  for (int g = 0; g < 8; ++g) { mrun[g] = -INFINITY; lrun[g] = 0.f; }
#pragma unroll
  for (int t = 0; t < 4; ++t)
#pragma unroll
    for (int g = 0; g < 8; ++g) o[t][g] = 0.f;

  for (int j = 0; j < 32; ++j) {
    const int kb = j * 32;

    __syncthreads();                      // kt reuse guard
    // ---- stage 4KB K chunk (contiguous) into LDS --------------------------
#if HAVE_ASYNC_LDS
#pragma unroll
    for (int u = 0; u < 2; ++u) {
      const int c = tid * 2 + u;          // 256 chunks of 16B
      async_copy16(Kh + (size_t)kb * 64 + c * 8, kt + c * 8);
    }
    wait_async0();
#else
#pragma unroll
    for (int u = 0; u < 2; ++u) {
      const int c = tid * 2 + u;
      *(v8bf*)(kt + c * 8) = *(const v8bf*)(Kh + (size_t)kb * 64 + c * 8);
    }
#endif
    __syncthreads();

    if (j < 31) {                         // prefetch next K/V chunks
      __builtin_prefetch(Kh + (size_t)(kb + 32) * 64 + tid * 16, 0, 0);
      __builtin_prefetch(VT + (size_t)(tid >> 1) * 1024 + kb + 32, 0, 0);
    }

    // ---- scores S = Q @ K_chunk^T (two 16-key groups) ---------------------
    v8f s[2];
#pragma unroll
    for (int cg = 0; cg < 2; ++cg) {
      const int krow = cg * 16 + l16;
      v16bf b0 = *(const v16bf*)(kt + krow * 64 + kb16);        // d 0..31
      v16bf b1 = *(const v16bf*)(kt + krow * 64 + 32 + kb16);   // d 32..63
      v8f c;
#pragma unroll
      for (int g = 0; g < 8; ++g) c[g] = 0.f;
      c = wmma_bf16(qa0.v, b0, c);
      c = wmma_bf16(qa1.v, b1, c);
      s[cg] = c;
    }

    // ---- masks + online softmax ------------------------------------------
    const int   lk0 = kb + l16, lk1 = kb + 16 + l16;
    const float km0 = km[lk0], km1 = km[lk1];
    const float qm0 = qm[lk0], qm1 = qm[lk1];
#pragma unroll
    for (int g = 0; g < 8; ++g) {
      const int lq = tq * 16 + g + half8;
      float s0 = s[0][g] * 0.125f;
      float s1 = s[1][g] * 0.125f;
      if (km0 == 0.f || lk0 > lq) s0 = NEGC;
      if (km1 == 0.f || lk1 > lq) s1 = NEGC;
      float cm = fmaxf(s0, s1);
#pragma unroll
      for (int m = 8; m >= 1; m >>= 1) cm = fmaxf(cm, __shfl_xor(cm, m, 32));
      const float nm = fmaxf(mrun[g], cm);
      const float alpha = __expf(mrun[g] - nm);
      const float p0 = __expf(s0 - nm);
      const float p1 = __expf(s1 - nm);
      float rs = p0 + p1;
#pragma unroll
      for (int m = 8; m >= 1; m >>= 1) rs += __shfl_xor(rs, m, 32);
      lrun[g] = lrun[g] * alpha + rs;
      mrun[g] = nm;
#pragma unroll
      for (int t = 0; t < 4; ++t) o[t][g] *= alpha;
      pt[wid][(g + half8) * 32 + l16]      = f2bf(p0 * qm0);   // qmask post-softmax
      pt[wid][(g + half8) * 32 + 16 + l16] = f2bf(p1 * qm1);
    }

    asm volatile("" ::: "memory");        // LDS in-order within wave

    // ---- O += P @ V_chunk (V columns contiguous in vT) --------------------
    Frag pa;
    pa.h[0] = *(const v8bf*)(&pt[wid][l16 * 32 + kb16]);
    pa.h[1] = *(const v8bf*)(&pt[wid][l16 * 32 + kb16 + 8]);
#pragma unroll
    for (int t = 0; t < 4; ++t) {
      v16bf vb = *(const v16bf*)(VT + (size_t)(t * 16 + l16) * 1024 + kb + kb16);
      o[t] = wmma_bf16(pa.v, vb, o[t]);
    }
  }

#pragma unroll
  for (int g = 0; g < 8; ++g) {
    const float inv = 1.f / fmaxf(lrun[g], 1e-30f);
    const int lq = tq * 16 + g + half8;
#pragma unroll
    for (int t = 0; t < 4; ++t)
      Oacc[(size_t)i * L * D + (size_t)lq * D + t * 16 + l16] = o[t][g] * inv;
  }
}

// ---------------------------------------------------------------------------
// P4: concat heads + residual + L2 normalize. One block (256 thr) per (b,l).
// ---------------------------------------------------------------------------
__global__ void k_finalize(const float* __restrict__ attnS, const float* __restrict__ query,
                           float* __restrict__ out) {
  const int row = blockIdx.x;
  const int b = row >> 10, l = row & 1023;
  const int tid = threadIdx.x;
  float vals[4], ss = 0.f;
#pragma unroll
  for (int k = 0; k < 4; ++k) {
    const int c = tid + k * 256;
    const int h = c >> 6, d = c & 63;
    float v = attnS[(((size_t)(h * 4 + b)) * 1024 + l) * 64 + d] +
              query[(size_t)row * 1024 + c];
    vals[k] = v;
    ss += v * v;
  }
#pragma unroll
  for (int m = 16; m >= 1; m >>= 1) ss += __shfl_xor(ss, m, 32);
  __shared__ float red[8];
  if ((tid & 31) == 0) red[tid >> 5] = ss;
  __syncthreads();
  float tot = 0.f;
#pragma unroll
  for (int w = 0; w < 8; ++w) tot += red[w];
  const float inv = 1.f / fmaxf(sqrtf(tot), 1e-12f);
#pragma unroll
  for (int k = 0; k < 4; ++k)
    out[(size_t)row * 1024 + tid + k * 256] = vals[k] * inv;
}

// ---------------------------------------------------------------------------
// host-side orchestration
// ---------------------------------------------------------------------------
extern "C" void kernel_launch(void* const* d_in, const int* in_sizes, int n_in,
                              void* d_out, int out_size, void* d_ws, size_t ws_size,
                              hipStream_t stream) {
  const float* query = (const float*)d_in[0];
  const float* key   = (const float*)d_in[1];
  const float* Wq    = (const float*)d_in[2];
  const float* Wk    = (const float*)d_in[3];
  const float* Wv    = (const float*)d_in[4];
  float* out = (float*)d_out;

  char* ws = (char*)d_ws;
  const size_t MB = 1024 * 1024;
  __bf16* qb    = (__bf16*)(ws + 0 * MB);    // 8 MB
  __bf16* kb    = (__bf16*)(ws + 8 * MB);    // 8 MB
  __bf16* WqT   = (__bf16*)(ws + 16 * MB);   // 2 MB (transposed bf16)
  __bf16* WkT   = (__bf16*)(ws + 18 * MB);
  __bf16* WvT   = (__bf16*)(ws + 20 * MB);
  __bf16* qp    = (__bf16*)(ws + 22 * MB);   // 8 MB
  __bf16* kp    = (__bf16*)(ws + 30 * MB);   // 8 MB
  __bf16* vT    = (__bf16*)(ws + 38 * MB);   // 8 MB (per-slice transposed V)
  float*  attnS = (float*)(ws + 46 * MB);    // 16 MB
  float*  kmask = (float*)(ws + 62 * MB);
  float*  qmask = (float*)(ws + 62 * MB + 16384);

  k_tobf16<<<4096, 256, 0, stream>>>(query, qb, 4 * 1024 * 1024);
  k_tobf16<<<4096, 256, 0, stream>>>(key,   kb, 4 * 1024 * 1024);

  dim3 gt(16, 16);
  k_wtrans<<<gt, 256, 0, stream>>>(Wq, WqT);
  k_wtrans<<<gt, 256, 0, stream>>>(Wk, WkT);
  k_wtrans<<<gt, 256, 0, stream>>>(Wv, WvT);

  k_rowmask<<<512, 256, 0, stream>>>(key,   kmask, 4096);
  k_rowmask<<<512, 256, 0, stream>>>(query, qmask, 4096);

  dim3 gg(16, 64);
  k_gemm_proj<<<gg, 128, 0, stream>>>(qb, WqT, qp, 0);
  k_gemm_proj<<<gg, 128, 0, stream>>>(kb, WkT, kp, 0);
  k_gemm_proj<<<gg, 128, 0, stream>>>(kb, WvT, vT, 1);

  dim3 ga(16, 64);
  k_attn<<<ga, 128, 0, stream>>>(qp, kp, vT, kmask, qmask, attnS);

  k_finalize<<<4096, 256, 0, stream>>>(attnS, query, out);
}
